// GATPolicy_79190607004136
// MI455X (gfx1250) — matrix operand order repaired and checked by reading, not compiled, not run on hardware
//
#include <hip/hip_runtime.h>
#include <hip/hip_bf16.h>

#define NEG_SLOPE 0.2f

typedef __attribute__((ext_vector_type(16))) _Float16 v16h;
typedef __attribute__((ext_vector_type(8)))  _Float16 h8;
typedef __attribute__((ext_vector_type(8)))  float    v8f;

// ---------------------------------------------------------------------------
// small helpers
// ---------------------------------------------------------------------------
__device__ __forceinline__ void atomicMaxF(float* addr, float v) {
    // classic monotone-bit-pattern trick; addr must be initialized to -inf
    if (v >= 0.0f) atomicMax((int*)addr, __float_as_int(v));
    else           atomicMin((unsigned int*)addr, __float_as_uint(v));
}

__device__ __forceinline__ float leaky(float v) {
    return v > 0.0f ? v : NEG_SLOPE * v;
}

__global__ void fill_u32_kernel(unsigned int* __restrict__ p, unsigned int v, int n) {
    int i = blockIdx.x * blockDim.x + threadIdx.x;
    if (i < n) p[i] = v;
}

// y[node][k] = (k < K) ? (f16)x[node][k] : 0   — pads activation rows to Kpad
__global__ void pad_convert_kernel(const float* __restrict__ x, _Float16* __restrict__ y,
                                   int K, int Kpad, int total /* = N*Kpad */) {
    int i = blockIdx.x * blockDim.x + threadIdx.x;
    if (i >= total) return;
    int node = i / Kpad, k = i % Kpad;
    y[i] = (k < K) ? (_Float16)x[(size_t)node * K + k] : (_Float16)0.0f;
}

// Wt[n][k] = (k < K) ? W[k][n] : 0   (transpose + zero-pad K up to Kpad)
__global__ void prep_wt_kernel(const float* __restrict__ W, _Float16* __restrict__ Wt,
                               int K, int Kpad, int Nout) {
    int i = blockIdx.x * blockDim.x + threadIdx.x;
    if (i >= Nout * Kpad) return;
    int nn = i / Kpad, k = i % Kpad;
    Wt[i] = (k < K) ? (_Float16)W[(size_t)k * Nout + nn] : (_Float16)0.0f;
}

// ---------------------------------------------------------------------------
// WMMA GEMM: C[M][NT*16] = A[M][KPAD](f16, zero-padded) * Bt[NT*16][KPAD]^T
// One wave per 16-row strip; A fragment reused across all NT column tiles.
// Fully unrolled K loop (KPAD compile-time); no divergent guards -> EXEC
// stays all-ones as WMMA requires.
// ---------------------------------------------------------------------------
template <int KPAD, int NT>
__global__ __launch_bounds__(256) void gemm_wmma_kernel(
    const _Float16* __restrict__ A, const _Float16* __restrict__ Bt,
    float* __restrict__ C, int M)
{
    constexpr int Nout = NT * 16;
    const int lane = threadIdx.x & 31;
    const int wave = threadIdx.x >> 5;
    const int gw   = blockIdx.x * 8 + wave;       // one 16-row strip per wave
    if (gw >= (M >> 4)) return;                   // wave-uniform exit
    const int mBase = gw << 4;
    const int llo = lane & 15;
    const int lhi = lane >> 4;

    v8f acc[NT];
#pragma unroll
    for (int nt = 0; nt < NT; ++nt)
#pragma unroll
        for (int i = 0; i < 8; ++i) acc[nt][i] = 0.0f;

    const _Float16* Arow = A + (size_t)(mBase + llo) * KPAD;

#pragma unroll
    for (int k0 = 0; k0 < KPAD; k0 += 32) {
        // A fragment (16x32 f16): lane row m=llo; halfs 0..7 @ k0+lhi*8, 8..15 @ +16
        h8 a0 = *(const h8*)(Arow + k0 + lhi * 8);
        h8 a1 = *(const h8*)(Arow + k0 + lhi * 8 + 16);
        v16h a;
#pragma unroll
        for (int i = 0; i < 8; ++i) { a[i] = a0[i]; a[8 + i] = a1[i]; }

#pragma unroll
        for (int nt = 0; nt < NT; ++nt) {
            // B fragment (32x16 f16): lane col n=llo; halfs 0..7 @ k0+lhi*8, 8..15 @ +16
            const _Float16* Brow = Bt + (size_t)(nt * 16 + llo) * KPAD + k0 + lhi * 8;
            h8 b0 = *(const h8*)(Brow);
            h8 b1 = *(const h8*)(Brow + 16);
            v16h b;
#pragma unroll
            for (int i = 0; i < 8; ++i) { b[i] = b0[i]; b[8 + i] = b1[i]; }
            acc[nt] = __builtin_amdgcn_wmma_f32_16x16x32_f16(
                          false, a, false, b, (short)0, acc[nt], false, false);
        }
    }

    // C/D layout: lane col n=llo; VGPR r -> row r + lhi*8
#pragma unroll
    for (int nt = 0; nt < NT; ++nt) {
        float* Cp = C + (size_t)(mBase + lhi * 8) * Nout + nt * 16 + llo;
#pragma unroll
        for (int r = 0; r < 8; ++r) Cp[(size_t)r * Nout] = acc[nt][r];
    }
}

// ---------------------------------------------------------------------------
// GAT pieces
// ---------------------------------------------------------------------------
__global__ void alpha_kernel(const float* __restrict__ hmat,
                             const float* __restrict__ a_src,
                             const float* __restrict__ a_dst,
                             float* __restrict__ as, float* __restrict__ ad,
                             int n, int Hl, int Cl) {
    int t = blockIdx.x * blockDim.x + threadIdx.x;
    if (t >= n * Hl) return;
    int node = t / Hl, hh = t % Hl;
    const float* hp = hmat + (size_t)node * Hl * Cl + (size_t)hh * Cl;
    float s1 = 0.0f, s2 = 0.0f;
    for (int c = 0; c < Cl; ++c) {
        float v = hp[c];
        s1 += v * a_src[hh * Cl + c];
        s2 += v * a_dst[hh * Cl + c];
    }
    as[t] = s1;
    ad[t] = s2;
}

__device__ __forceinline__ void edge_decode(int e, int Ee, const int* src, const int* dst,
                                            int& s, int& d) {
    if (e < Ee) { s = src[e]; d = dst[e]; }
    else        { s = d = e - Ee; }        // self loop
}

__global__ void edge_max_kernel(const int* __restrict__ src, const int* __restrict__ dst,
                                const float* __restrict__ as, const float* __restrict__ ad,
                                float* __restrict__ emax, int Ee, int n, int Hl) {
    int t = blockIdx.x * blockDim.x + threadIdx.x;
    if (t >= (Ee + n) * Hl) return;
    int e = t / Hl, hh = t % Hl, s, d;
    edge_decode(e, Ee, src, dst, s, d);
    float v = leaky(as[s * Hl + hh] + ad[d * Hl + hh]);
    atomicMaxF(&emax[d * Hl + hh], v);
}

__global__ void edge_sum_kernel(const int* __restrict__ src, const int* __restrict__ dst,
                                const float* __restrict__ as, const float* __restrict__ ad,
                                const float* __restrict__ emax, float* __restrict__ den,
                                int Ee, int n, int Hl) {
    int t = blockIdx.x * blockDim.x + threadIdx.x;
    if (t >= (Ee + n) * Hl) return;
    int e = t / Hl, hh = t % Hl, s, d;
    edge_decode(e, Ee, src, dst, s, d);
    int di = d * Hl + hh;
    float v  = leaky(as[s * Hl + hh] + ad[di]);
    float ex = expf(v - emax[di]);
    atomicAdd(&den[di], ex);
}

__global__ void edge_scatter_kernel(const int* __restrict__ src, const int* __restrict__ dst,
                                    const float* __restrict__ as, const float* __restrict__ ad,
                                    const float* __restrict__ emax, const float* __restrict__ den,
                                    const float* __restrict__ hmat, float* __restrict__ acc,
                                    int Ee, int n, int Hl, int Cl) {
    int t = blockIdx.x * blockDim.x + threadIdx.x;
    if (t >= (Ee + n) * Hl) return;
    int e = t / Hl, hh = t % Hl, s, d;
    edge_decode(e, Ee, src, dst, s, d);
    int di = d * Hl + hh;
    float v     = leaky(as[s * Hl + hh] + ad[di]);
    float ex    = expf(v - emax[di]);
    float alpha = ex / (den[di] + 1e-16f);
    const float* hp = hmat + (size_t)s * Hl * Cl + (size_t)hh * Cl;
    float*       ap = acc  + (size_t)d * Hl * Cl + (size_t)hh * Cl;
    for (int c = 0; c < Cl; ++c) atomicAdd(&ap[c], hp[c] * alpha);
}

// x[i] = act(x[i] + b[i % width]);  act = ELU or identity
__global__ void bias_act_kernel(float* __restrict__ x, const float* __restrict__ b,
                                int total, int width, int doElu) {
    int i = blockIdx.x * blockDim.x + threadIdx.x;
    if (i >= total) return;
    float v = x[i] + b[i % width];
    x[i] = doElu ? (v > 0.0f ? v : expf(v) - 1.0f) : v;
}

// ---------------------------------------------------------------------------
// pooling
// ---------------------------------------------------------------------------
__global__ void count_kernel(const int* __restrict__ bat, float* __restrict__ cnt, int n) {
    int i = blockIdx.x * blockDim.x + threadIdx.x;
    if (i < n) atomicAdd(&cnt[bat[i]], 1.0f);
}

__global__ void pool_sum_kernel(const float* __restrict__ x2, const int* __restrict__ bat,
                                float* __restrict__ pooled, int n, int Cl) {
    int i = blockIdx.x * blockDim.x + threadIdx.x;
    if (i >= n * Cl) return;
    int node = i / Cl, c = i % Cl;
    atomicAdd(&pooled[bat[node] * Cl + c], x2[i]);
}

__global__ void pool_div_kernel(float* __restrict__ pooled, const float* __restrict__ cnt,
                                int total, int Cl) {
    int i = blockIdx.x * blockDim.x + threadIdx.x;
    if (i >= total) return;
    pooled[i] /= fmaxf(cnt[i / Cl], 1.0f);
}

// ---------------------------------------------------------------------------
// MLP head: one block (256 threads) per batch row; shared-mem LayerNorm
// out layout: logits[64][16] then value[64]
// ---------------------------------------------------------------------------
__global__ __launch_bounds__(256) void head_kernel(
    const float* __restrict__ pooled, const float* __restrict__ obs,
    const float* __restrict__ Ws1, const float* __restrict__ bs1,
    const float* __restrict__ lng, const float* __restrict__ lnb,
    const float* __restrict__ Ws2, const float* __restrict__ bs2,
    const float* __restrict__ Wa,  const float* __restrict__ ba,
    const float* __restrict__ Wc,  const float* __restrict__ bc,
    float* __restrict__ out)
{
    __shared__ float comb[192];
    __shared__ float h1[256];
    __shared__ float h2[256];
    __shared__ float red[256];
    const int b = blockIdx.x;
    const int t = threadIdx.x;

    if (t < 192) comb[t] = (t < 64) ? pooled[b * 64 + t] : obs[b * 128 + (t - 64)];
    __syncthreads();

    float h = bs1[t];
    for (int k = 0; k < 192; ++k) h += comb[k] * Ws1[k * 256 + t];

    // mean
    red[t] = h;
    __syncthreads();
    for (int s = 128; s > 0; s >>= 1) {
        if (t < s) red[t] += red[t + s];
        __syncthreads();
    }
    float mu = red[0] / 256.0f;
    __syncthreads();

    // variance
    float dmu = h - mu;
    red[t] = dmu * dmu;
    __syncthreads();
    for (int s = 128; s > 0; s >>= 1) {
        if (t < s) red[t] += red[t + s];
        __syncthreads();
    }
    float var = red[0] / 256.0f;
    __syncthreads();

    float hn = dmu * rsqrtf(var + 1e-5f) * lng[t] + lnb[t];
    h1[t] = fmaxf(hn, 0.0f);
    __syncthreads();

    float a2 = bs2[t];
    for (int k = 0; k < 256; ++k) a2 += h1[k] * Ws2[k * 256 + t];
    h2[t] = fmaxf(a2, 0.0f);
    __syncthreads();

    if (t < 16) {
        float lg = ba[t];
        for (int k = 0; k < 256; ++k) lg += h2[k] * Wa[k * 16 + t];
        out[b * 16 + t] = lg;
    }
    if (t == 16) {
        float v = bc[0];
        for (int k = 0; k < 256; ++k) v += h2[k] * Wc[k];
        out[64 * 16 + b] = v;
    }
}

// ---------------------------------------------------------------------------
// per-layer driver (KPAD/NT compile-time so the GEMM fully unrolls)
// ---------------------------------------------------------------------------
struct LayerBufs {
    float* X32a; float* X32b; _Float16* XH; _Float16* WT;
    float* AS; float* AD; float* EM; float* DEN;
    const int* srcI; const int* dstI;
};

template <int KPAD, int NT>
static void run_layer(const LayerBufs& B, const float* xin, int K, int Hl, int Cl,
                      const float* W, const float* asrc, const float* adst,
                      const float* bvec, int doElu, hipStream_t stream) {
    constexpr int Nn = 50000, Ee = 800000;
    const int EL = Ee + Nn;
    const int NoutL = Hl * Cl;
    const int T = 256;
    auto blk = [&](int n) { return (n + T - 1) / T; };

    // f16 zero-padded activations (after this, xin/X32a is dead -> reusable as acc)
    pad_convert_kernel<<<blk(Nn * KPAD), T, 0, stream>>>(xin, B.XH, K, KPAD, Nn * KPAD);
    // f16 transposed + K-padded weights
    prep_wt_kernel<<<blk(NoutL * KPAD), T, 0, stream>>>(W, B.WT, K, KPAD, NoutL);
    // h = x @ W  (WMMA), one wave per 16-row strip
    const int strips = Nn / 16;
    gemm_wmma_kernel<KPAD, NT><<<(strips + 7) / 8, T, 0, stream>>>(B.XH, B.WT, B.X32b, Nn);
    // attention logits per node
    alpha_kernel<<<blk(Nn * Hl), T, 0, stream>>>(B.X32b, asrc, adst, B.AS, B.AD, Nn, Hl, Cl);
    // init segment buffers
    fill_u32_kernel<<<blk(Nn * Hl), T, 0, stream>>>((unsigned*)B.EM, 0xFF800000u, Nn * Hl);
    fill_u32_kernel<<<blk(Nn * Hl), T, 0, stream>>>((unsigned*)B.DEN, 0u, Nn * Hl);
    fill_u32_kernel<<<blk(Nn * NoutL), T, 0, stream>>>((unsigned*)B.X32a, 0u, Nn * NoutL);
    // scatter-softmax + aggregation
    const int etot = EL * Hl;
    edge_max_kernel<<<blk(etot), T, 0, stream>>>(B.srcI, B.dstI, B.AS, B.AD, B.EM, Ee, Nn, Hl);
    edge_sum_kernel<<<blk(etot), T, 0, stream>>>(B.srcI, B.dstI, B.AS, B.AD, B.EM, B.DEN,
                                                 Ee, Nn, Hl);
    edge_scatter_kernel<<<blk(etot), T, 0, stream>>>(B.srcI, B.dstI, B.AS, B.AD, B.EM, B.DEN,
                                                     B.X32b, B.X32a, Ee, Nn, Hl, Cl);
    // + bias, optional ELU
    bias_act_kernel<<<blk(Nn * NoutL), T, 0, stream>>>(B.X32a, bvec, Nn * NoutL, NoutL, doElu);
}

// ---------------------------------------------------------------------------
// launch
// ---------------------------------------------------------------------------
extern "C" void kernel_launch(void* const* d_in, const int* in_sizes, int n_in,
                              void* d_out, int out_size, void* d_ws, size_t ws_size,
                              hipStream_t stream) {
    (void)in_sizes; (void)n_in; (void)out_size; (void)ws_size;
    constexpr int Nn = 50000, Bb = 64;
    constexpr int F0v = 8, Hh = 2, Cc = 64, HC = 128;
    constexpr int Ee = 800000;

    const float* obs = (const float*)d_in[0];
    const float* nf  = (const float*)d_in[1];
    const int*   ei  = (const int*)  d_in[2];
    const int*   bat = (const int*)  d_in[3];
    const float* W0  = (const float*)d_in[4];
    const float* as0 = (const float*)d_in[5];
    const float* ad0 = (const float*)d_in[6];
    const float* b0  = (const float*)d_in[7];
    const float* W1  = (const float*)d_in[8];
    const float* as1 = (const float*)d_in[9];
    const float* ad1 = (const float*)d_in[10];
    const float* b1  = (const float*)d_in[11];
    const float* W2  = (const float*)d_in[12];
    const float* as2 = (const float*)d_in[13];
    const float* ad2 = (const float*)d_in[14];
    const float* b2  = (const float*)d_in[15];
    const float* Ws1 = (const float*)d_in[16];
    const float* bs1 = (const float*)d_in[17];
    const float* lng = (const float*)d_in[18];
    const float* lnb = (const float*)d_in[19];
    const float* Ws2 = (const float*)d_in[20];
    const float* bs2 = (const float*)d_in[21];
    const float* Wa  = (const float*)d_in[22];
    const float* ba  = (const float*)d_in[23];
    const float* Wc  = (const float*)d_in[24];
    const float* bc  = (const float*)d_in[25];

    // workspace carve (256-byte aligned)
    char* ws = (char*)d_ws;
    size_t off = 0;
    auto carve = [&](size_t bytes) -> char* {
        char* r = ws + off;
        off += (bytes + 255) & ~(size_t)255;
        return r;
    };
    LayerBufs B;
    B.X32a = (float*)   carve((size_t)Nn * HC * sizeof(float));     // x / acc (ping)
    B.X32b = (float*)   carve((size_t)Nn * HC * sizeof(float));     // h = x@W
    B.XH   = (_Float16*)carve((size_t)Nn * HC * sizeof(_Float16));  // f16 padded input
    B.WT   = (_Float16*)carve((size_t)HC * HC * sizeof(_Float16));  // f16 W^T (padded)
    B.AS   = (float*)   carve((size_t)Nn * Hh * sizeof(float));
    B.AD   = (float*)   carve((size_t)Nn * Hh * sizeof(float));
    B.EM   = (float*)   carve((size_t)Nn * Hh * sizeof(float));
    B.DEN  = (float*)   carve((size_t)Nn * Hh * sizeof(float));
    float* POOL = (float*)carve((size_t)Bb * Cc * sizeof(float));
    float* CNT  = (float*)carve((size_t)Bb * sizeof(float));
    B.srcI = ei;
    B.dstI = ei + Ee;

    const int T = 256;
    auto blk = [&](int n) { return (n + T - 1) / T; };

    run_layer<32, 8>(B, nf,     F0v, Hh, Cc, W0, as0, ad0, b0, 1, stream); // 8 -> 128, ELU
    run_layer<128, 8>(B, B.X32a, HC, Hh, Cc, W1, as1, ad1, b1, 1, stream); // 128 -> 128, ELU
    run_layer<128, 4>(B, B.X32a, HC, 1,  Cc, W2, as2, ad2, b2, 0, stream); // 128 -> 64, no act

    // global mean pool
    fill_u32_kernel<<<blk(Bb * Cc), T, 0, stream>>>((unsigned*)POOL, 0u, Bb * Cc);
    fill_u32_kernel<<<1, T, 0, stream>>>((unsigned*)CNT, 0u, Bb);
    count_kernel<<<blk(Nn), T, 0, stream>>>(bat, CNT, Nn);
    pool_sum_kernel<<<blk(Nn * Cc), T, 0, stream>>>(B.X32a, bat, POOL, Nn, Cc);
    pool_div_kernel<<<blk(Bb * Cc), T, 0, stream>>>(POOL, CNT, Bb * Cc, Cc);

    // MLP head -> d_out (logits then values)
    head_kernel<<<Bb, T, 0, stream>>>(POOL, obs, Ws1, bs1, lng, lnb,
                                      Ws2, bs2, Wa, ba, Wc, bc, (float*)d_out);
}
